// NeuMF_50835232916081
// MI455X (gfx1250) — matrix operand order repaired and checked by reading, not compiled, not run on hardware
//
#include <hip/hip_runtime.h>

typedef __attribute__((ext_vector_type(16))) _Float16 v16h;
typedef __attribute__((ext_vector_type(8)))  _Float16 v8h;
typedef __attribute__((ext_vector_type(8)))  float    v8f;
typedef __attribute__((ext_vector_type(4)))  float    f32x4;

#define WAVES_PER_BLOCK 8
#define ROWS_PER_WAVE   16
#define GRID_BLOCKS     1024

// ReLU pinned to a single v_max_num_f32 (inline asm so instcombine cannot
// re-expand it into canonicalize + max).
__device__ __forceinline__ float relu1(float x) {
    float r;
    asm("v_max_num_f32 %0, %1, 0" : "=v"(r) : "v"(x));
    return r;
}

__global__ __launch_bounds__(256) void neumf_wmma_kernel(
    const int*   __restrict__ user_idx,
    const int*   __restrict__ item_idx,
    const float* __restrict__ emb_user_mlp,
    const float* __restrict__ emb_item_mlp,
    const float* __restrict__ emb_user_mf,
    const float* __restrict__ emb_item_mf,
    const float* __restrict__ W0, const float* __restrict__ B0,
    const float* __restrict__ W1, const float* __restrict__ B1,
    const float* __restrict__ W2, const float* __restrict__ B2,
    const float* __restrict__ Wo, const float* __restrict__ Bo,
    float* __restrict__ out, int batch)
{
    // Per-wave 1KB LDS staging buffer for D->A relayout between layers.
    // Views: 16x32 f16 (h0), 16x16 f16 (h1), 16x16 f32 (h2, cols 8-15 junk).
    __shared__ __align__(16) _Float16 smem[WAVES_PER_BLOCK * 16 * 32];

    const int lane = threadIdx.x & 31;
    const int wave = threadIdx.x >> 5;
    const int hs   = lane >> 4;    // 0: lanes 0-15, 1: lanes 16-31
    const int n    = lane & 15;    // column index (B/D) == row-in-tile (A)
    const int k0   = hs * 8;

    _Float16* hb = smem + wave * (16 * 32);
    float*    fb = (float*)hb;

    // ================= loop-invariant operand construction (once per wave) ===
    // B layout (32x16 f16), column N=n: element j<8 holds K=8*hs+j,
    // element j>=8 holds K=16+8*hs+(j-8) (zero where K is padded past the real depth).
    v16h b0a, b0b, b1v, b2v;
    #pragma unroll
    for (int j = 0; j < 16; ++j) {
        float wa = (j < 8) ? W0[(8 * hs + j) * 32 + n]      : 0.f;    // layer0, N=0..15
        float wb = (j < 8) ? W0[(8 * hs + j) * 32 + n + 16] : 0.f;    // layer0, N=16..31
        b0a[j] = (_Float16)wa;
        b0b[j] = (_Float16)wb;
        b1v[j] = (_Float16)W1[((j < 8) ? (8 * hs + j) : (8 + 8 * hs + j)) * 16 + n];       // layer1, full K=32
        float w2 = (j < 8 && n < 8) ? W2[(8 * hs + j) * 8 + n] : 0.f; // layer2, N<8, K<16
        b2v[j] = (_Float16)w2;
    }

    // Bias accumulators (C operands), replicated across the 8 M-rows per lane.
    v8f c0a, c0b, c1, c2;
    {
        float ba = B0[n], bb = B0[n + 16], b1s = B1[n];
        float b2s = (n < 8) ? B2[n] : 0.f;
        #pragma unroll
        for (int r = 0; r < 8; ++r) {
            c0a[r] = ba; c0b[r] = bb; c1[r] = b1s; c2[r] = b2s;
        }
    }

    // Output affine weights (lane-uniform -> scalar loads).
    float wo[16];
    #pragma unroll
    for (int j = 0; j < 16; ++j) wo[j] = Wo[j];
    const float bos = Bo[0];

    // ================= grid-stride loop over 16-row batch tiles ==============
    const int nTiles     = (batch + ROWS_PER_WAVE - 1) / ROWS_PER_WAVE;
    const int tileStride = gridDim.x * WAVES_PER_BLOCK;

    for (int tile = blockIdx.x * WAVES_PER_BLOCK + wave; tile < nTiles;
         tile += tileStride) {
        const int base = tile * ROWS_PER_WAVE;
        const int row  = base + n;
        const int rowc = row < batch ? row : batch - 1;

        // ---------------- embedding gathers ----------------
        const int uidx = user_idx[rowc];
        const int iidx = item_idx[rowc];

        // A0: row m=n of X = concat(u_mlp, i_mlp). Lanes 0-15 hold K=0..7
        // (user emb), lanes 16-31 hold K=8..15 (item emb). K=16..31 zero.
        const float* mlp_src = (hs == 0) ? (emb_user_mlp + (size_t)uidx * 8)
                                         : (emb_item_mlp + (size_t)iidx * 8);
        f32x4 e0 = *(const f32x4*)(mlp_src);
        f32x4 e1 = *(const f32x4*)(mlp_src + 4);

        // GMF branch: both lane halves fetch the same addresses; requests
        // dedup to the same cachelines (tables live in the 192MB L2).
        const float* um = emb_user_mf + (size_t)uidx * 8;
        const float* im = emb_item_mf + (size_t)iidx * 8;
        f32x4 u0 = *(const f32x4*)um, u1 = *(const f32x4*)(um + 4);
        f32x4 i0 = *(const f32x4*)im, i1 = *(const f32x4*)(im + 4);
        float mfv[8];
        mfv[0] = u0.x * i0.x; mfv[1] = u0.y * i0.y;
        mfv[2] = u0.z * i0.z; mfv[3] = u0.w * i0.w;
        mfv[4] = u1.x * i1.x; mfv[5] = u1.y * i1.y;
        mfv[6] = u1.z * i1.z; mfv[7] = u1.w * i1.w;

        // ---------------- layer 0: [16,16] @ [16,32] ----------------
        v16h a0;
        #pragma unroll
        for (int j = 0; j < 16; ++j) a0[j] = (_Float16)0.f;
        a0[0] = (_Float16)e0.x; a0[1] = (_Float16)e0.y;
        a0[2] = (_Float16)e0.z; a0[3] = (_Float16)e0.w;
        a0[4] = (_Float16)e1.x; a0[5] = (_Float16)e1.y;
        a0[6] = (_Float16)e1.z; a0[7] = (_Float16)e1.w;

        v8f d0a = __builtin_amdgcn_wmma_f32_16x16x32_f16(
            false, a0, false, b0a, (short)0, c0a, false, false);
        v8f d0b = __builtin_amdgcn_wmma_f32_16x16x32_f16(
            false, a0, false, b0b, (short)0, c0b, false, false);

        // ReLU (one v_max_num_f32 each) + stage h0 (16x32 f16) into LDS.
        // D layout: VGPR r -> M = r + 8*hs, N = n (+16 for second tile).
        #pragma unroll
        for (int r = 0; r < 8; ++r) {
            int M = r + 8 * hs;
            hb[M * 32 + n]      = (_Float16)relu1(d0a[r]);
            hb[M * 32 + n + 16] = (_Float16)relu1(d0b[r]);
        }

        // ---------------- layer 1: [16,32] @ [32,16] ----------------
        // A layout load: lane holds row m=n; halfs 0..7 = K=k0..k0+7,
        // halfs 8..15 = K=16+k0..16+k0+7 (two ds_load_b128).
        v8h l1lo = *(const v8h*)&hb[n * 32 + k0];
        v8h l1hi = *(const v8h*)&hb[n * 32 + 16 + k0];
        v16h a1;
        #pragma unroll
        for (int j = 0; j < 8; ++j) { a1[j] = l1lo[j]; a1[j + 8] = l1hi[j]; }

        v8f d1 = __builtin_amdgcn_wmma_f32_16x16x32_f16(
            false, a1, false, b1v, (short)0, c1, false, false);

        // ReLU + stage h1 (16x16 f16, row stride 16).
        #pragma unroll
        for (int r = 0; r < 8; ++r) {
            int M = r + 8 * hs;
            hb[M * 16 + n] = (_Float16)relu1(d1[r]);
        }

        // ---------------- layer 2: [16,16] @ [16,8] ----------------
        v8h l2lo = *(const v8h*)&hb[n * 16 + k0];
        v16h a2;
        #pragma unroll
        for (int j = 0; j < 8; ++j) { a2[j] = l2lo[j]; a2[j + 8] = (_Float16)0.f; }

        v8f d2 = __builtin_amdgcn_wmma_f32_16x16x32_f16(
            false, a2, false, b2v, (short)0, c2, false, false);

        // ReLU + stage h2 as 16x16 f32 (exactly 1KB). All 32 lanes store
        // unconditionally (cols 8-15 are junk, never read) -> no EXEC
        // masking around the ds stores. DS is in-order within a wave, so
        // the a2 loads above read before these stores overwrite.
        #pragma unroll
        for (int r = 0; r < 8; ++r) {
            fb[(r + 8 * hs) * 16 + n] = relu1(d2[r]);
        }

        // ---------------- fusion + affine output (lanes 0-15) ----------------
        if (hs == 0 && row < batch) {
            f32x4 h2a = *(const f32x4*)&fb[n * 16];
            f32x4 h2b = *(const f32x4*)&fb[n * 16 + 4];
            float acc = bos;
            acc += h2a.x * wo[0] + h2a.y * wo[1] + h2a.z * wo[2] + h2a.w * wo[3];
            acc += h2b.x * wo[4] + h2b.y * wo[5] + h2b.z * wo[6] + h2b.w * wo[7];
            #pragma unroll
            for (int k = 0; k < 8; ++k) acc += mfv[k] * wo[8 + k];
            out[row] = acc;
        }
    }
}

extern "C" void kernel_launch(void* const* d_in, const int* in_sizes, int n_in,
                              void* d_out, int out_size, void* d_ws, size_t ws_size,
                              hipStream_t stream) {
    const int*   user_idx     = (const int*)d_in[0];
    const int*   item_idx     = (const int*)d_in[1];
    const float* emb_user_mlp = (const float*)d_in[2];
    const float* emb_item_mlp = (const float*)d_in[3];
    const float* emb_user_mf  = (const float*)d_in[4];
    const float* emb_item_mf  = (const float*)d_in[5];
    const float* W0 = (const float*)d_in[6];
    const float* B0 = (const float*)d_in[7];
    const float* W1 = (const float*)d_in[8];
    const float* B1 = (const float*)d_in[9];
    const float* W2 = (const float*)d_in[10];
    const float* B2 = (const float*)d_in[11];
    const float* Wo = (const float*)d_in[12];
    const float* Bo = (const float*)d_in[13];
    float* out = (float*)d_out;

    const int batch  = in_sizes[0];
    const int nTiles = (batch + ROWS_PER_WAVE - 1) / ROWS_PER_WAVE;
    int blocks = (nTiles + WAVES_PER_BLOCK - 1) / WAVES_PER_BLOCK;
    if (blocks > GRID_BLOCKS) blocks = GRID_BLOCKS;

    neumf_wmma_kernel<<<blocks, WAVES_PER_BLOCK * 32, 0, stream>>>(
        user_idx, item_idx,
        emb_user_mlp, emb_item_mlp, emb_user_mf, emb_item_mf,
        W0, B0, W1, B1, W2, B2, Wo, Bo, out, batch);
}
